// SwinEncoderBlock_87033217286284
// MI455X (gfx1250) — matrix operand order repaired and checked by reading, not compiled, not run on hardware
//
#include <hip/hip_runtime.h>
#include <math.h>
#include <stdint.h>

// ---------------------------------------------------------------------------
// Types for CDNA5 WMMA (wave32, 16x16x32 f16 -> f32 accum)
// ---------------------------------------------------------------------------
typedef _Float16 half_t;
typedef __attribute__((ext_vector_type(16))) _Float16 v16h;
typedef __attribute__((ext_vector_type(8)))  _Float16 v8h;
typedef __attribute__((ext_vector_type(8)))  float    v8f;

constexpr int kB = 2, kT = 4, kH = 64, kW = 64;
constexpr int kDIM = 512, kHEADS = 16, kWS = 8, kSHIFT = 4;
constexpr int kHD = kDIM / kHEADS;           // 32
constexpr int kL = kWS * kWS;                // 64
constexpr int kMLP = 4 * kDIM;               // 2048
constexpr int kNT = kB * kT * kH * kW;       // 32768 tokens
constexpr int kNWIN = kB * kT * (kH/kWS) * (kW/kWS); // 512 windows
constexpr float kEPS = 1e-5f;

// ---------------------------------------------------------------------------
// Tensor Data Mover (TDM) support, guarded for toolchain availability
// ---------------------------------------------------------------------------
#if defined(__has_builtin)
#  if __has_builtin(__builtin_amdgcn_tensor_load_to_lds) && \
      __has_builtin(__builtin_amdgcn_s_wait_tensorcnt)
#    define USE_TDM 1
#  endif
#endif
#ifndef USE_TDM
#  define USE_TDM 0
#endif

__device__ __forceinline__ unsigned lds_off_u32(const void* p) {
  // LDS generic addresses map to the LDS window by dropping upper bits
  return (unsigned)(uintptr_t)p;
}

#if USE_TDM
typedef unsigned int v4u  __attribute__((ext_vector_type(4)));
typedef int          v8i_t __attribute__((ext_vector_type(8)));
typedef int          v4i_t __attribute__((ext_vector_type(4)));

// Issue one 2D f16 tile DMA: global (row-major, row_stride_elems) -> LDS with
// hardware padding (pad codes per D# group1: interval 0=2dw..., amount 0=1dw...).
// D# layout per CDNA5 ISA ch.8 (group0 128b, group1 256b), groups 2/3 unused.
__device__ __forceinline__ void tdm_load_tile_f16(
    const half_t* gsrc, unsigned lds_addr,
    unsigned tile_cols, unsigned tile_rows, unsigned row_stride_elems,
    unsigned pad_interval_code, unsigned pad_amount_code) {
  const unsigned long long ga = (unsigned long long)(uintptr_t)gsrc;
  v4u g0;
  g0.x = 1u;                                                 // count=1, user D#
  g0.y = lds_addr;                                           // lds_addr[31:0]
  g0.z = (unsigned)ga;                                       // global_addr[31:0]
  g0.w = (unsigned)((ga >> 32) & 0x01FFFFFFu) | (2u << 30);  // addr[56:32]|type=2
  v8i_t g1;
  g1[0] = (int)((1u << 16) |                                 // data_size = 2B
                (1u << 20) |                                 // pad_enable
                (pad_interval_code << 22) |
                (pad_amount_code << 25));
  g1[1] = (int)((row_stride_elems & 0xFFFFu) << 16);         // tensor_dim0[15:0]
  g1[2] = (int)(((row_stride_elems >> 16) & 0xFFFFu) |       // tensor_dim0[31:16]
                ((tile_rows & 0xFFFFu) << 16));              // tensor_dim1[15:0]
  g1[3] = (int)(((tile_rows >> 16) & 0xFFFFu) |              // tensor_dim1[31:16]
                ((tile_cols & 0xFFFFu) << 16));              // tile_dim0
  g1[4] = (int)(tile_rows & 0xFFFFu);                        // tile_dim1 (dim2=0)
  g1[5] = (int)row_stride_elems;                             // dim0_stride[31:0]
  g1[6] = 0;                                                 // stride hi / dim1_stride
  g1[7] = 0;
  const v4i_t zero4 = {0, 0, 0, 0};
#if __clang_major__ >= 23
  typedef int v8z __attribute__((ext_vector_type(8)));
  const v8z zero8 = {0, 0, 0, 0, 0, 0, 0, 0};
  __builtin_amdgcn_tensor_load_to_lds(g0, g1, zero4, zero4, zero8, 0);
#else
  __builtin_amdgcn_tensor_load_to_lds(g0, g1, zero4, zero4, 0);
#endif
}
#endif  // USE_TDM

// ---------------------------------------------------------------------------
// WMMA fragment helpers (A/B in contiguous k order k = 16*(lane/16) + s, wave32)
// ---------------------------------------------------------------------------
__device__ __forceinline__ v16h lds_load_a(const half_t* p, int ld, int lane) {
  const int row = lane & 15;
  const int kb  = (lane >> 4) << 4;     // 0 or 16
  const half_t* r = p + row * ld;
  v16h a;
#pragma unroll
  for (int h2 = 0; h2 < 8; ++h2) a[h2] = r[kb + h2];
#pragma unroll
  for (int h2 = 0; h2 < 8; ++h2) a[8 + h2] = r[kb + 8 + h2];
  return a;
}
__device__ __forceinline__ v16h lds_load_b(const half_t* p, int ld, int lane) {
  const int col = lane & 15;
  const int kb  = (lane >> 4) << 4;     // 0 or 16
  v16h b;
#pragma unroll
  for (int h2 = 0; h2 < 16; ++h2) b[h2] = p[(kb + h2) * ld + col];
  return b;
}
__device__ __forceinline__ v16h lds_load_bt(const half_t* p, int ld, int lane) {
  const int col = lane & 15;
  const int kb  = (lane >> 4) << 4;
  const half_t* r = p + col * ld + kb;
  v16h b;
#pragma unroll
  for (int h2 = 0; h2 < 16; ++h2) b[h2] = r[h2];
  return b;
}
__device__ __forceinline__ v8f wmma_f32_16x16x32(v16h a, v16h b, v8f c) {
  return __builtin_amdgcn_wmma_f32_16x16x32_f16(false, a, false, b, (short)0, c,
                                                false, false);
}

// ---------------------------------------------------------------------------
// f32 -> f16 weight conversion
// ---------------------------------------------------------------------------
__global__ __launch_bounds__(256) void cvt_f16_kernel(const float* __restrict__ src,
                                                      half_t* __restrict__ dst, int n) {
  int i = blockIdx.x * blockDim.x + threadIdx.x;
  if (i < n) dst[i] = (half_t)src[i];
}

// ---------------------------------------------------------------------------
// LN1 + cyclic shift (-SHIFT,-SHIFT) + window partition (gather, f16 out)
// ---------------------------------------------------------------------------
__global__ __launch_bounds__(256) void ln1_shift_window_kernel(
    const float* __restrict__ x, const float* __restrict__ w,
    const float* __restrict__ b, half_t* __restrict__ yw) {
  __shared__ float red[256];
  const int r = blockIdx.x;
  const int win = r >> 6, l = r & 63;
  const int bm = win >> 6, wh = (win >> 3) & 7, ww = win & 7;
  const int ih = l >> 3, iw = l & 7;
  const int hh = (wh * kWS + ih + kSHIFT) & (kH - 1);
  const int wc = (ww * kWS + iw + kSHIFT) & (kW - 1);
  const size_t token = (size_t)bm * (kH * kW) + hh * kW + wc;
  const float* xr = x + token * kDIM;
  const int tid = threadIdx.x;
  const float v0 = xr[tid], v1 = xr[tid + 256];
  red[tid] = v0 + v1;
  __syncthreads();
  for (int s = 128; s > 0; s >>= 1) { if (tid < s) red[tid] += red[tid + s]; __syncthreads(); }
  const float mean = red[0] * (1.0f / kDIM);
  __syncthreads();
  red[tid] = v0 * v0 + v1 * v1;
  __syncthreads();
  for (int s = 128; s > 0; s >>= 1) { if (tid < s) red[tid] += red[tid + s]; __syncthreads(); }
  const float var = red[0] * (1.0f / kDIM) - mean * mean;
  const float rstd = rsqrtf(var + kEPS);
  half_t* yr = yw + (size_t)r * kDIM;
  yr[tid]       = (half_t)((v0 - mean) * rstd * w[tid] + b[tid]);
  yr[tid + 256] = (half_t)((v1 - mean) * rstd * w[tid + 256] + b[tid + 256]);
}

// ---------------------------------------------------------------------------
// Generic WMMA GEMM: C[M,N] = act(A[M,K] @ B[K,N] + bias) (+ resid + gamma)
// Block tile 128x128, 256 threads = 8 waves (2x4), wave tile 64x32.
// TDM path: double-buffered LDS, tiles staged by the Tensor Data Mover.
// ---------------------------------------------------------------------------
template <typename OutT, bool GELU, bool RESID>
__global__ __launch_bounds__(256) void gemm_wmma_kernel(
    const half_t* __restrict__ A, const half_t* __restrict__ B,
    const float* __restrict__ bias, const float* __restrict__ resid,
    const float* __restrict__ gamma, OutT* __restrict__ C,
    int M, int N, int K) {
  __shared__ __align__(16) half_t As[2][128][40];   // 80B rows (16B aligned)
  __shared__ __align__(16) half_t Bs[2][32][136];   // 272B rows (16B aligned)
  const int tid = threadIdx.x;
  const int lane = tid & 31;
  const int wv = tid >> 5;
  const int wm = wv >> 2, wn = wv & 3;
  const int m0 = blockIdx.y * 128, n0 = blockIdx.x * 128;
  const int row0 = wm * 64, col0 = wn * 32;

  v8f acc[4][2] = {};

#if USE_TDM
  // Prologue: wave 0 launches tile 0 DMA for both operands.
  if (wv == 0) {
    tdm_load_tile_f16(&A[(size_t)m0 * K], lds_off_u32(&As[0][0][0]),
                      32, 128, (unsigned)K, /*16dw*/3, /*4dw*/3);
    tdm_load_tile_f16(&B[(size_t)0 * N + n0], lds_off_u32(&Bs[0][0][0]),
                      128, 32, (unsigned)N, /*64dw*/5, /*4dw*/3);
  }
  int cur = 0;
  for (int k0 = 0; k0 < K; k0 += 32) {
    if (wv == 0) __builtin_amdgcn_s_wait_tensorcnt(0);
    __syncthreads();                       // tile `cur` visible to all waves
    if (wv == 0 && (k0 + 32) < K) {        // overlap next tile DMA with math
      tdm_load_tile_f16(&A[(size_t)m0 * K + (k0 + 32)],
                        lds_off_u32(&As[cur ^ 1][0][0]), 32, 128, (unsigned)K, 3, 3);
      tdm_load_tile_f16(&B[(size_t)(k0 + 32) * N + n0],
                        lds_off_u32(&Bs[cur ^ 1][0][0]), 128, 32, (unsigned)N, 5, 3);
    }
    v16h af[4];
#pragma unroll
    for (int mt = 0; mt < 4; ++mt)
      af[mt] = lds_load_a(&As[cur][row0 + mt * 16][0], 40, lane);
    v16h bf[2];
#pragma unroll
    for (int nt = 0; nt < 2; ++nt)
      bf[nt] = lds_load_b(&Bs[cur][0][col0 + nt * 16], 136, lane);
#pragma unroll
    for (int mt = 0; mt < 4; ++mt)
#pragma unroll
      for (int nt = 0; nt < 2; ++nt)
        acc[mt][nt] = wmma_f32_16x16x32(af[mt], bf[nt], acc[mt][nt]);
    cur ^= 1;
  }
#else
  for (int k0 = 0; k0 < K; k0 += 32) {
#pragma unroll
    for (int it = 0; it < 2; ++it) {
      const int c = tid + it * 256;
      const int ar = c >> 2, ac = (c & 3) << 3;
      *(v8h*)&As[0][ar][ac] = *(const v8h*)&A[(size_t)(m0 + ar) * K + k0 + ac];
      const int br = c >> 4, bc = (c & 15) << 3;
      *(v8h*)&Bs[0][br][bc] = *(const v8h*)&B[(size_t)(k0 + br) * N + n0 + bc];
      if (k0 + 32 < K) {
        __builtin_prefetch(&A[(size_t)(m0 + ar) * K + k0 + 32 + ac], 0, 3);
        __builtin_prefetch(&B[(size_t)(k0 + 32 + br) * N + n0 + bc], 0, 3);
      }
    }
    __syncthreads();
    v16h af[4];
#pragma unroll
    for (int mt = 0; mt < 4; ++mt)
      af[mt] = lds_load_a(&As[0][row0 + mt * 16][0], 40, lane);
    v16h bf[2];
#pragma unroll
    for (int nt = 0; nt < 2; ++nt)
      bf[nt] = lds_load_b(&Bs[0][0][col0 + nt * 16], 136, lane);
#pragma unroll
    for (int mt = 0; mt < 4; ++mt)
#pragma unroll
      for (int nt = 0; nt < 2; ++nt)
        acc[mt][nt] = wmma_f32_16x16x32(af[mt], bf[nt], acc[mt][nt]);
    __syncthreads();
  }
#endif

  const int rb = (lane >> 4) << 3;
  const int cl = lane & 15;
#pragma unroll
  for (int mt = 0; mt < 4; ++mt)
#pragma unroll
    for (int nt = 0; nt < 2; ++nt)
#pragma unroll
      for (int i = 0; i < 8; ++i) {
        const int row = m0 + row0 + mt * 16 + rb + i;
        const int col = n0 + col0 + nt * 16 + cl;
        float val = acc[mt][nt][i] + bias[col];
        if (GELU) val = 0.5f * val * (1.0f + erff(val * 0.7071067811865476f));
        if (RESID) val = resid[(size_t)row * N + col] + gamma[0] * val;
        C[(size_t)row * N + col] = (OutT)val;
      }
}

// ---------------------------------------------------------------------------
// Windowed attention: one block per (window, head); 128 threads = 4 waves.
// ---------------------------------------------------------------------------
__global__ __launch_bounds__(128) void swin_attn_kernel(
    const half_t* __restrict__ q, const half_t* __restrict__ k,
    const half_t* __restrict__ v, const float* __restrict__ bias_table,
    half_t* __restrict__ out) {
  __shared__ __align__(16) half_t Qs[64][40];
  __shared__ __align__(16) half_t Ks[64][40];
  __shared__ __align__(16) half_t Vs[64][40];
  __shared__ float Ss[64][65];
  __shared__ __align__(16) half_t Ps[64][72];
  const int blk = blockIdx.x;
  const int win = blk >> 4, head = blk & 15;
  const int tid = threadIdx.x;
  const size_t base = (size_t)win * kL * kDIM + head * kHD;

#if USE_TDM
  if (tid < 32) {  // wave 0 launches the three [64,32] tile DMAs
    tdm_load_tile_f16(q + base, lds_off_u32(&Qs[0][0]), 32, 64, kDIM, 3, 3);
    tdm_load_tile_f16(k + base, lds_off_u32(&Ks[0][0]), 32, 64, kDIM, 3, 3);
    tdm_load_tile_f16(v + base, lds_off_u32(&Vs[0][0]), 32, 64, kDIM, 3, 3);
    __builtin_amdgcn_s_wait_tensorcnt(0);
  }
  __syncthreads();
#else
  for (int c = tid; c < 256; c += 128) {
    const int row = c >> 2, cc = (c & 3) << 3;
    const size_t gi = base + (size_t)row * kDIM + cc;
    *(v8h*)&Qs[row][cc] = *(const v8h*)&q[gi];
    *(v8h*)&Ks[row][cc] = *(const v8h*)&k[gi];
    *(v8h*)&Vs[row][cc] = *(const v8h*)&v[gi];
  }
  __syncthreads();
#endif

  const int wv = tid >> 5, lane = tid & 31;
  const int row0 = wv << 4;
  const int rb = (lane >> 4) << 3, cl = lane & 15;
  const float scale = 0.17677669529663687f; // 1/sqrt(HD)

  const v16h qa = lds_load_a(&Qs[row0][0], 40, lane);
#pragma unroll
  for (int mt = 0; mt < 4; ++mt) {
    const v16h kb2 = lds_load_bt(&Ks[mt * 16][0], 40, lane);
    v8f s = {};
    s = wmma_f32_16x16x32(qa, kb2, s);
    const int m = mt * 16 + cl;
#pragma unroll
    for (int i = 0; i < 8; ++i) {
      const int lrow = row0 + rb + i;
      const int relh = (lrow >> 3) - (m >> 3) + (kWS - 1);
      const int relw = (lrow & 7) - (m & 7) + (kWS - 1);
      const float bv = bias_table[(relh * (2 * kWS - 1) + relw) * kHEADS + head];
      Ss[lrow][m] = s[i] * scale + bv;
    }
  }
  __syncthreads();

  if (tid < 64) {
    float mx = -1e30f;
    for (int j = 0; j < 64; ++j) mx = fmaxf(mx, Ss[tid][j]);
    float ssum = 0.0f;
    for (int j = 0; j < 64; ++j) { float e = __expf(Ss[tid][j] - mx); ssum += e; Ss[tid][j] = e; }
    const float inv = 1.0f / ssum;
    for (int j = 0; j < 64; ++j) Ps[tid][j] = (half_t)(Ss[tid][j] * inv);
  }
  __syncthreads();

  const v16h pa0 = lds_load_a(&Ps[row0][0], 72, lane);
  const v16h pa1 = lds_load_a(&Ps[row0][32], 72, lane);
#pragma unroll
  for (int nt = 0; nt < 2; ++nt) {
    v8f o = {};
    o = wmma_f32_16x16x32(pa0, lds_load_b(&Vs[0][nt * 16], 40, lane), o);
    o = wmma_f32_16x16x32(pa1, lds_load_b(&Vs[32][nt * 16], 40, lane), o);
#pragma unroll
    for (int i = 0; i < 8; ++i) {
      const size_t orow = (size_t)win * kL + row0 + rb + i;
      out[orow * kDIM + head * kHD + nt * 16 + cl] = (half_t)o[i];
    }
  }
}

// ---------------------------------------------------------------------------
// Window-reverse + un-shift + x = 2x + gamma_attn*y, then LN2 -> f16
// ---------------------------------------------------------------------------
__global__ __launch_bounds__(256) void resid_ln2_kernel(
    const float* __restrict__ x, const float* __restrict__ attnproj,
    const float* __restrict__ gamma_attn, const float* __restrict__ w,
    const float* __restrict__ b, float* __restrict__ x1,
    half_t* __restrict__ hout) {
  __shared__ float red[256];
  const int n = blockIdx.x;
  const int bm = n >> 12, hh = (n >> 6) & 63, wc = n & 63;
  const int hs = (hh - kSHIFT) & 63, wsx = (wc - kSHIFT) & 63;
  const int wh = hs >> 3, ih = hs & 7, ww = wsx >> 3, iw = wsx & 7;
  const size_t r = (size_t)(((bm * 8 + wh) * 8 + ww) * 64 + ih * 8 + iw);
  const int tid = threadIdx.x;
  const float g = gamma_attn[0];
  const float* xr = x + (size_t)n * kDIM;
  const float* ar = attnproj + r * kDIM;
  const float u0 = 2.0f * xr[tid]       + g * ar[tid];
  const float u1 = 2.0f * xr[tid + 256] + g * ar[tid + 256];
  float* x1r = x1 + (size_t)n * kDIM;
  x1r[tid] = u0;
  x1r[tid + 256] = u1;
  red[tid] = u0 + u1;
  __syncthreads();
  for (int s = 128; s > 0; s >>= 1) { if (tid < s) red[tid] += red[tid + s]; __syncthreads(); }
  const float mean = red[0] * (1.0f / kDIM);
  __syncthreads();
  red[tid] = u0 * u0 + u1 * u1;
  __syncthreads();
  for (int s = 128; s > 0; s >>= 1) { if (tid < s) red[tid] += red[tid + s]; __syncthreads(); }
  const float var = red[0] * (1.0f / kDIM) - mean * mean;
  const float rstd = rsqrtf(var + kEPS);
  half_t* hr = hout + (size_t)n * kDIM;
  hr[tid]       = (half_t)((u0 - mean) * rstd * w[tid] + b[tid]);
  hr[tid + 256] = (half_t)((u1 - mean) * rstd * w[tid + 256] + b[tid + 256]);
}

// ---------------------------------------------------------------------------
// Host-side orchestration
// ---------------------------------------------------------------------------
extern "C" void kernel_launch(void* const* d_in, const int* in_sizes, int n_in,
                              void* d_out, int out_size, void* d_ws, size_t ws_size,
                              hipStream_t stream) {
  (void)in_sizes; (void)n_in; (void)out_size; (void)ws_size;
  const float* x     = (const float*)d_in[0];
  const float* n1w   = (const float*)d_in[1];
  const float* n1b   = (const float*)d_in[2];
  const float* qw    = (const float*)d_in[3];
  const float* qb    = (const float*)d_in[4];
  const float* kw    = (const float*)d_in[5];
  const float* kbv   = (const float*)d_in[6];
  const float* vw    = (const float*)d_in[7];
  const float* vb    = (const float*)d_in[8];
  const float* ow    = (const float*)d_in[9];
  const float* ob    = (const float*)d_in[10];
  const float* btab  = (const float*)d_in[11];
  const float* gattn = (const float*)d_in[12];
  const float* n2w   = (const float*)d_in[13];
  const float* n2b   = (const float*)d_in[14];
  const float* w1    = (const float*)d_in[15];
  const float* b1    = (const float*)d_in[16];
  const float* w2    = (const float*)d_in[17];
  const float* b2    = (const float*)d_in[18];
  const float* gmlp  = (const float*)d_in[19];

  size_t off = 0;
  char* base = (char*)d_ws;
  auto wsalloc = [&](size_t bytes) -> void* {
    void* p = base + off;
    off = (off + bytes + 255) & ~(size_t)255;
    return p;
  };
  half_t* yw  = (half_t*)wsalloc((size_t)kNT * kDIM * 2);
  half_t* qh  = (half_t*)wsalloc((size_t)kNT * kDIM * 2);
  half_t* kh  = (half_t*)wsalloc((size_t)kNT * kDIM * 2);
  half_t* vh  = (half_t*)wsalloc((size_t)kNT * kDIM * 2);
  half_t* ah  = (half_t*)wsalloc((size_t)kNT * kDIM * 2);
  float*  tmp = (float*) wsalloc((size_t)kNT * kDIM * 4);
  float*  x1  = (float*) wsalloc((size_t)kNT * kDIM * 4);
  half_t* hh  = (half_t*)wsalloc((size_t)kNT * kDIM * 2);
  half_t* m1  = (half_t*)wsalloc((size_t)kNT * kMLP * 2);
  half_t* qwf = (half_t*)wsalloc((size_t)kDIM * kDIM * 2);
  half_t* kwf = (half_t*)wsalloc((size_t)kDIM * kDIM * 2);
  half_t* vwf = (half_t*)wsalloc((size_t)kDIM * kDIM * 2);
  half_t* owf = (half_t*)wsalloc((size_t)kDIM * kDIM * 2);
  half_t* w1f = (half_t*)wsalloc((size_t)kDIM * kMLP * 2);
  half_t* w2f = (half_t*)wsalloc((size_t)kMLP * kDIM * 2);

  const int nsq = kDIM * kDIM, nw1 = kDIM * kMLP;
  cvt_f16_kernel<<<(nsq + 255) / 256, 256, 0, stream>>>(qw, qwf, nsq);
  cvt_f16_kernel<<<(nsq + 255) / 256, 256, 0, stream>>>(kw, kwf, nsq);
  cvt_f16_kernel<<<(nsq + 255) / 256, 256, 0, stream>>>(vw, vwf, nsq);
  cvt_f16_kernel<<<(nsq + 255) / 256, 256, 0, stream>>>(ow, owf, nsq);
  cvt_f16_kernel<<<(nw1 + 255) / 256, 256, 0, stream>>>(w1, w1f, nw1);
  cvt_f16_kernel<<<(nw1 + 255) / 256, 256, 0, stream>>>(w2, w2f, nw1);

  ln1_shift_window_kernel<<<kNT, 256, 0, stream>>>(x, n1w, n1b, yw);

  const dim3 gproj(kDIM / 128, kNT / 128);   // (4, 256)
  const dim3 gfc1(kMLP / 128, kNT / 128);    // (16, 256)

  gemm_wmma_kernel<half_t, false, false><<<gproj, 256, 0, stream>>>(
      yw, qwf, qb, nullptr, nullptr, qh, kNT, kDIM, kDIM);
  gemm_wmma_kernel<half_t, false, false><<<gproj, 256, 0, stream>>>(
      yw, kwf, kbv, nullptr, nullptr, kh, kNT, kDIM, kDIM);
  gemm_wmma_kernel<half_t, false, false><<<gproj, 256, 0, stream>>>(
      yw, vwf, vb, nullptr, nullptr, vh, kNT, kDIM, kDIM);

  swin_attn_kernel<<<kNWIN * kHEADS, 128, 0, stream>>>(qh, kh, vh, btab, ah);

  gemm_wmma_kernel<float, false, false><<<gproj, 256, 0, stream>>>(
      ah, owf, ob, nullptr, nullptr, tmp, kNT, kDIM, kDIM);

  resid_ln2_kernel<<<kNT, 256, 0, stream>>>(x, tmp, gattn, n2w, n2b, x1, hh);

  gemm_wmma_kernel<half_t, true, false><<<gfc1, 256, 0, stream>>>(
      hh, w1f, b1, nullptr, nullptr, m1, kNT, kMLP, kDIM);

  gemm_wmma_kernel<float, false, true><<<gproj, 256, 0, stream>>>(
      m1, w2f, b2, x1, gmlp, (float*)d_out, kNT, kDIM, kMLP);
}
